// GINRegressor_17617955848276
// MI455X (gfx1250) — matrix operand order repaired and checked by reading, not compiled, not run on hardware
//
#include <hip/hip_runtime.h>

// ---------------------------------------------------------------------------
// GIN regressor for MI455X (gfx1250), wave32, WMMA bf16 (fp32 accumulate).
// Edge pipeline (the ~2.5 GB mover) is fused: WMMA ee + gather + ReLU +
// atomic scatter.  Weights are pre-packed into WMMA B-fragment order so each
// lane loads its operands as one contiguous 32B vector.
// ---------------------------------------------------------------------------

typedef __bf16 bf16;
typedef __attribute__((ext_vector_type(16))) __bf16 v16bf;
typedef __attribute__((ext_vector_type(8)))  float  v8f;

#define N_NODES 50000
#define N_EDGES 800000
#define IN_CH   64
#define HID     128
#define EDGE_DIM 16
#define NLAYERS 3
#define NGRAPH  256
#define JKDIM   (HID * NLAYERS)   // 384
#define HEADK   (JKDIM * 2)       // 768

__device__ __forceinline__ bf16 f2bf(float f) {
  // round-to-nearest-even fp32 -> bf16
  unsigned u = __builtin_bit_cast(unsigned, f);
  u += 0x7fffu + ((u >> 16) & 1u);
  unsigned short h = (unsigned short)(u >> 16);
  return __builtin_bit_cast(bf16, h);
}

// ------------------------------- utility kernels ---------------------------

__global__ __launch_bounds__(256) void k_zero(float* __restrict__ p, int n) {
  int i = blockIdx.x * 256 + threadIdx.x;
  if (i < n) p[i] = 0.0f;
}

// Pack a [K,128] fp32 weight into WMMA B-fragment order (bf16), K padded up
// to a multiple of 32 with zeros.  Fragment f = kt*8 + nt occupies 32 lanes x
// 16 elements; lane reads its 16 operands as one contiguous 32B chunk.
__global__ __launch_bounds__(256) void k_packB(const float* __restrict__ s,
                                               bf16* __restrict__ d,
                                               int Kreal, int n) {
  int i = blockIdx.x * 256 + threadIdx.x;
  if (i >= n) return;
  const int j    = i & 15;
  const int lane = (i >> 4) & 31;
  const int t    = i >> 9;          // fragment id = kt*8 + nt
  const int nt   = t & 7;
  const int kt   = t >> 3;
  const int k    = kt * 32 + ((lane >= 16) ? 8 : 0) + (j & 7) + ((j >= 8) ? 16 : 0);   // ISA B 32x16 layout
  const int col  = nt * 16 + (lane & 15);
  d[i] = (k < Kreal) ? f2bf(s[k * HID + col]) : f2bf(0.0f);
}

// ------------------------------- generic WMMA GEMM -------------------------
// C[M,128] = op(A)[M,K] @ B[K,128] + bias ; optional per-column BN stats.
// Block = 256 threads = 8 waves; block handles 16 rows, wave w -> cols 16w..16w+15.
// MODE 0: A as-is | MODE 1: A + A2 ([M,128]) | MODE 2: relu(A*scl[k]+shf[k])
template <int MODE, bool STATS>
__global__ __launch_bounds__(256) void k_gemm(
    const float* __restrict__ A, int lda,
    const float* __restrict__ A2,
    const float* __restrict__ scl, const float* __restrict__ shf,
    const bf16* __restrict__ Bp,          // packed fragments
    const float* __restrict__ bias,
    float* __restrict__ C,
    float* __restrict__ ssum, float* __restrict__ ssq,
    int K)
{
  const int lane  = threadIdx.x & 31;
  const int ntile = threadIdx.x >> 5;              // 0..7
  const bool hi   = lane >= 16;
  const int row   = blockIdx.x * 16 + (lane & 15); // A row fed by this lane
  const int col   = ntile * 16 + (lane & 15);      // C col owned by this lane
  const int kbA   = hi ? 8 : 0;

  v8f c = {};                                      // bias added in epilogue
  const int ktiles = K >> 5;
  for (int kt = 0; kt < ktiles; ++kt) {
    const int k0 = kt * 32;
    // A fragment (ISA 16-bit A 16x32): two contiguous 32B runs per lane.
    float av[16];
    {
      const float* ap = A + row * lda + k0 + kbA;
      *(float4*)(av + 0)  = *(const float4*)(ap + 0);
      *(float4*)(av + 4)  = *(const float4*)(ap + 4);
      *(float4*)(av + 8)  = *(const float4*)(ap + 16);
      *(float4*)(av + 12) = *(const float4*)(ap + 20);
    }
    if (MODE == 1) {
      float bv[16];
      const float* ap2 = A2 + row * HID + k0 + kbA;
      *(float4*)(bv + 0)  = *(const float4*)(ap2 + 0);
      *(float4*)(bv + 4)  = *(const float4*)(ap2 + 4);
      *(float4*)(bv + 8)  = *(const float4*)(ap2 + 16);
      *(float4*)(bv + 12) = *(const float4*)(ap2 + 20);
#pragma unroll
      for (int j = 0; j < 16; ++j) av[j] += bv[j];
    }
    if (MODE == 2) {
      float sv[16], tv[16];
      const float* sp = scl + k0 + kbA;
      const float* tp = shf + k0 + kbA;
      *(float4*)(sv + 0)  = *(const float4*)(sp + 0);
      *(float4*)(sv + 4)  = *(const float4*)(sp + 4);
      *(float4*)(sv + 8)  = *(const float4*)(sp + 16);
      *(float4*)(sv + 12) = *(const float4*)(sp + 20);
      *(float4*)(tv + 0)  = *(const float4*)(tp + 0);
      *(float4*)(tv + 4)  = *(const float4*)(tp + 4);
      *(float4*)(tv + 8)  = *(const float4*)(tp + 16);
      *(float4*)(tv + 12) = *(const float4*)(tp + 20);
#pragma unroll
      for (int j = 0; j < 16; ++j) av[j] = fmaxf(av[j] * sv[j] + tv[j], 0.0f);
    }
    v16bf a;
#pragma unroll
    for (int j = 0; j < 16; ++j) a[j] = f2bf(av[j]);

    // B fragment: one contiguous 32B vector load from the packed layout.
    const v16bf b = *(const v16bf*)(Bp + (size_t)((kt * 8 + ntile) * 32 + lane) * 16);

    c = __builtin_amdgcn_wmma_f32_16x16x32_bf16(false, a, false, b,
                                                (short)0, c, false, false);
  }

  const int rbase = blockIdx.x * 16 + (hi ? 8 : 0);
  const float bi = bias[col];
  float s = 0.0f, q = 0.0f;
#pragma unroll
  for (int r = 0; r < 8; ++r) {
    const float v = c[r] + bi;
    C[(rbase + r) * HID + col] = v;
    s += v; q += v * v;
  }
  if (STATS) {
    atomicAdd(&ssum[col], s);
    atomicAdd(&ssq[col], q);
  }
}

// ------------------------------- fused GINE edge kernel --------------------
// Per wave: 16 edges.  ee = edge_attr@We (WMMA, K=16 zero-padded to 32),
// msg = relu(h[src] + ee + be), atomicAdd into agg[dst].
__global__ __launch_bounds__(256) void k_edge(
    const float* __restrict__ ea,       // [E,16]
    const bf16* __restrict__ Wb,        // packed [32x128] fragments (K padded)
    const float* __restrict__ be,       // [128]
    const float* __restrict__ h, int hstride,
    const int* __restrict__ src, const int* __restrict__ dst,
    float* __restrict__ agg)            // [N,128]
{
  const int lane = threadIdx.x & 31;
  const int wave = threadIdx.x >> 5;
  const bool hi  = lane >= 16;
  const int ebase = (blockIdx.x * 8 + wave) * 16;

  // A fragment: real K=0..15 (two 16B vector loads), K=16..31 zero-padded.
  v16bf a;
  {
    float av[8];
    const float* ap = ea + (size_t)(ebase + (lane & 15)) * EDGE_DIM + (hi ? 8 : 0);
    *(float4*)(av + 0) = *(const float4*)(ap + 0);
    *(float4*)(av + 4) = *(const float4*)(ap + 4);
#pragma unroll
    for (int j = 0; j < 8; ++j) a[j] = f2bf(av[j]);
    const bf16 z = f2bf(0.0f);
#pragma unroll
    for (int j = 8; j < 16; ++j) a[j] = z;
  }

  // Row base addresses for the 8 edge-rows this lane owns in the C fragment.
  int hbase[8], abase[8];
  {
    const int r0 = ebase + (hi ? 8 : 0);
#pragma unroll
    for (int r = 0; r < 8; ++r) {
      hbase[r] = src[r0 + r] * hstride;
      abase[r] = dst[r0 + r] * HID;
    }
  }

  const int cn = lane & 15;
  for (int nt = 0; nt < 8; ++nt) {
    const int col = nt * 16 + cn;
    const v16bf b = *(const v16bf*)(Wb + (size_t)(nt * 32 + lane) * 16);
    v8f c = {};
    c = __builtin_amdgcn_wmma_f32_16x16x32_bf16(false, a, false, b,
                                                (short)0, c, false, false);
    // Batch all 8 gathers so they are in flight together, then scatter.
    float hv[8];
#pragma unroll
    for (int r = 0; r < 8; ++r) hv[r] = h[hbase[r] + col];
    const float bi = be[col];
#pragma unroll
    for (int r = 0; r < 8; ++r) {
      const float m = fmaxf(c[r] + bi + hv[r], 0.0f);
      atomicAdd(&agg[abase[r] + col], m);
    }
  }
}

// ------------------------------- BN helpers --------------------------------

__global__ void k_bnfin(const float* __restrict__ sum, const float* __restrict__ sq,
                        const float* __restrict__ g, const float* __restrict__ b,
                        float invn, float* __restrict__ scl, float* __restrict__ shf)
{
  const int c = threadIdx.x;  // 128
  const float m   = sum[c] * invn;
  const float var = fmaxf(sq[c] * invn - m * m, 0.0f);
  const float s   = g[c] * rsqrtf(var + 1e-5f);
  scl[c] = s;
  shf[c] = b[c] - m * s;
}

// h_next = relu(v*scale+shift) written into the JK concat buffer (strided slice)
__global__ __launch_bounds__(256) void k_bnapply(
    const float* __restrict__ v, const float* __restrict__ scl,
    const float* __restrict__ shf, float* __restrict__ dst, int dstride)
{
  const int i = blockIdx.x * 256 + threadIdx.x;  // [0, N*128), exact
  const int row = i >> 7, c = i & 127;
  dst[row * dstride + c] = fmaxf(v[i] * scl[c] + shf[c], 0.0f);
}

// ------------------------------- pooling (no atomics) ----------------------
// batch is sorted: block g binary-searches its node range and reduces
// mean+max straight into the head GEMM's A matrix Ap[g] = [mean(384)|max(384)].
__global__ __launch_bounds__(384) void k_poolseg(
    const float* __restrict__ xj, const int* __restrict__ batch,
    float* __restrict__ Ap)             // [G, 768]
{
  const int g = blockIdx.x;    // 0..255
  const int c = threadIdx.x;   // 0..383
  int lo = 0, hb = N_NODES;
  while (lo < hb) { const int mid = (lo + hb) >> 1; if (batch[mid] < g) lo = mid + 1; else hb = mid; }
  const int s = lo;
  hb = N_NODES;
  while (lo < hb) { const int mid = (lo + hb) >> 1; if (batch[mid] < g + 1) lo = mid + 1; else hb = mid; }
  const int e = lo;
  float sum = 0.0f, mx = 0.0f;
  for (int r = s; r < e; ++r) {
    const float v = xj[(size_t)r * JKDIM + c];
    sum += v;
    mx = fmaxf(mx, v);              // xj is post-ReLU (>= 0)
  }
  const float inv = 1.0f / fmaxf((float)(e - s), 1.0f);
  Ap[g * HEADK + c] = sum * inv;
  Ap[g * HEADK + JKDIM + c] = mx;
}

// ------------------------------- head tail ---------------------------------

__global__ void k_head2(const float* __restrict__ hu, const float* __restrict__ scl,
                        const float* __restrict__ shf, const float* __restrict__ W2,
                        const float* __restrict__ b2, float* __restrict__ out)
{
  const int g = threadIdx.x;  // 256
  float acc = b2[0];
#pragma unroll 4
  for (int c = 0; c < HID; ++c)
    acc += fmaxf(hu[g * HID + c] * scl[c] + shf[c], 0.0f) * W2[c];
  out[g] = acc;
}

// ------------------------------- launcher ----------------------------------

extern "C" void kernel_launch(void* const* d_in, const int* in_sizes, int n_in,
                              void* d_out, int out_size, void* d_ws, size_t ws_size,
                              hipStream_t stream)
{
  (void)in_sizes; (void)n_in; (void)out_size; (void)ws_size;

  const float* x        = (const float*)d_in[0];
  const float* edge_attr= (const float*)d_in[1];
  const int*   src      = (const int*)  d_in[2];
  const int*   dstE     = (const int*)  d_in[3];
  const int*   batch    = (const int*)  d_in[4];
  const float* Wp       = (const float*)d_in[5];
  const float* bp       = (const float*)d_in[6];
  const float* conv_We  = (const float*)d_in[7];
  const float* conv_be  = (const float*)d_in[8];
  const float* conv_W1  = (const float*)d_in[9];
  const float* conv_b1  = (const float*)d_in[10];
  const float* conv_g1  = (const float*)d_in[11];
  const float* conv_bt1 = (const float*)d_in[12];
  const float* conv_W2  = (const float*)d_in[13];
  const float* conv_b2  = (const float*)d_in[14];
  const float* bn_g     = (const float*)d_in[15];
  const float* bn_b     = (const float*)d_in[16];
  const float* head_W1  = (const float*)d_in[17];
  const float* head_b1  = (const float*)d_in[18];
  const float* head_g   = (const float*)d_in[19];
  const float* head_bt  = (const float*)d_in[20];
  const float* head_W2  = (const float*)d_in[21];
  const float* head_b2  = (const float*)d_in[22];
  float* out = (float*)d_out;

  // ---- workspace carve-up (256B aligned) ----
  char* ws = (char*)d_ws;
  size_t off = 0;
  auto carve = [&](size_t bytes) -> char* {
    char* p = ws + off;
    off += (bytes + 255) & ~(size_t)255;
    return p;
  };
  float* h0    = (float*)carve((size_t)N_NODES * HID * 4);
  float* u     = (float*)carve((size_t)N_NODES * HID * 4);
  float* v     = (float*)carve((size_t)N_NODES * HID * 4);
  float* agg   = (float*)carve((size_t)N_NODES * HID * 4);
  float* xj    = (float*)carve((size_t)N_NODES * JKDIM * 4);
  float* Apool = (float*)carve((size_t)NGRAPH * HEADK * 4);
  float* headu = (float*)carve((size_t)NGRAPH * HID * 4);
  float* stats = (float*)carve(7 * 256 * 4);   // 7 BNs x (sum[128], sumsq[128])
  float* sshf  = (float*)carve(7 * 256 * 4);   // 7 BNs x (scale[128], shift[128])
  // packed bf16 weight fragments: (Kpad/32) * 4096 elements each
  bf16* Wpbf   = (bf16*)carve((size_t)2  * 4096 * 2);           // K=64
  bf16* Webf   = (bf16*)carve((size_t)NLAYERS * 4096 * 2);      // K=16 -> pad 32
  bf16* W1bf   = (bf16*)carve((size_t)NLAYERS * 4 * 4096 * 2);  // K=128
  bf16* W2bf   = (bf16*)carve((size_t)NLAYERS * 4 * 4096 * 2);  // K=128
  bf16* hW1bf  = (bf16*)carve((size_t)24 * 4096 * 2);           // K=768

  // ---- zero accumulators (every call: deterministic under graph replay) ----
  k_zero<<<7, 256, 0, stream>>>(stats, 7 * 256);

  // ---- pack weights into WMMA B-fragment order ----
  k_packB<<<(2 * 4096) / 256, 256, 0, stream>>>(Wp, Wpbf, IN_CH, 2 * 4096);
  for (int l = 0; l < NLAYERS; ++l) {
    k_packB<<<4096 / 256, 256, 0, stream>>>(
        conv_We + (size_t)l * EDGE_DIM * HID, Webf + (size_t)l * 4096, EDGE_DIM, 4096);
    k_packB<<<(4 * 4096) / 256, 256, 0, stream>>>(
        conv_W1 + (size_t)l * HID * HID, W1bf + (size_t)l * 4 * 4096, HID, 4 * 4096);
    k_packB<<<(4 * 4096) / 256, 256, 0, stream>>>(
        conv_W2 + (size_t)l * HID * HID, W2bf + (size_t)l * 4 * 4096, HID, 4 * 4096);
  }
  k_packB<<<(24 * 4096) / 256, 256, 0, stream>>>(head_W1, hW1bf, HEADK, 24 * 4096);

  const int gemmGrid = N_NODES / 16;  // 3125

  // ---- node projection: h0 = x @ Wp + bp ----
  k_gemm<0, false><<<gemmGrid, 256, 0, stream>>>(
      x, IN_CH, nullptr, nullptr, nullptr, Wpbf, bp, h0, nullptr, nullptr, IN_CH);

  const float* hprev = h0;
  int hstride = HID;

  for (int l = 0; l < NLAYERS; ++l) {
    float* st1 = stats + (2 * l) * 256;
    float* st2 = stats + (2 * l + 1) * 256;
    float* ss1 = sshf + (2 * l) * 256;
    float* ss2 = sshf + (2 * l + 1) * 256;

    // agg = 0 ; agg += relu(h[src] + edge_attr@We + be) scattered to dst
    k_zero<<<(N_NODES * HID) / 256, 256, 0, stream>>>(agg, N_NODES * HID);
    k_edge<<<N_EDGES / 128, 256, 0, stream>>>(
        edge_attr, Webf + (size_t)l * 4096, conv_be + l * HID,
        hprev, hstride, src, dstE, agg);

    // u = (h + agg) @ W1 + b1   (+ BN1 stats)
    k_gemm<1, true><<<gemmGrid, 256, 0, stream>>>(
        hprev, hstride, agg, nullptr, nullptr,
        W1bf + (size_t)l * 4 * 4096, conv_b1 + l * HID,
        u, st1, st1 + 128, HID);
    k_bnfin<<<1, 128, 0, stream>>>(st1, st1 + 128, conv_g1 + l * HID,
                                   conv_bt1 + l * HID, 1.0f / N_NODES, ss1, ss1 + 128);

    // v = relu(BN1(u)) @ W2 + b2   (+ BN2 stats)
    k_gemm<2, true><<<gemmGrid, 256, 0, stream>>>(
        u, HID, nullptr, ss1, ss1 + 128,
        W2bf + (size_t)l * 4 * 4096, conv_b2 + l * HID,
        v, st2, st2 + 128, HID);
    k_bnfin<<<1, 128, 0, stream>>>(st2, st2 + 128, bn_g + l * HID,
                                   bn_b + l * HID, 1.0f / N_NODES, ss2, ss2 + 128);

    // h = relu(BN2(v)) written straight into the JK concat slice
    k_bnapply<<<(N_NODES * HID) / 256, 256, 0, stream>>>(
        v, ss2, ss2 + 128, xj + l * HID, JKDIM);

    hprev = xj + l * HID;
    hstride = JKDIM;
  }

  // ---- graph pooling: segment mean + max, atomic-free (batch is sorted) ----
  k_poolseg<<<NGRAPH, 384, 0, stream>>>(xj, batch, Apool);

  // ---- head: z @ head_W1 + b1, BN, relu, @ head_W2 + b2 ----
  float* sth = stats + 6 * 256;
  float* ssh = sshf + 6 * 256;
  k_gemm<0, true><<<NGRAPH / 16, 256, 0, stream>>>(
      Apool, HEADK, nullptr, nullptr, nullptr, hW1bf, head_b1,
      headu, sth, sth + 128, HEADK);
  k_bnfin<<<1, 128, 0, stream>>>(sth, sth + 128, head_g, head_bt,
                                 1.0f / NGRAPH, ssh, ssh + 128);
  k_head2<<<1, 256, 0, stream>>>(headu, ssh, ssh + 128, head_W2, head_b2, out);
}
